// SelfAttention_10411000726225
// MI455X (gfx1250) — matrix operand order repaired and checked by reading, not compiled, not run on hardware
//
#include <hip/hip_runtime.h>
#include <hip/hip_bf16.h>
#include <math.h>
#include <stdint.h>

typedef __bf16 bf16;
typedef __attribute__((ext_vector_type(16))) __bf16 v16bf;
typedef __attribute__((ext_vector_type(8)))  __bf16 v8bf;
typedef __attribute__((ext_vector_type(4)))  __bf16 v4bf;
typedef __attribute__((ext_vector_type(8)))  float  v8f;

#define WMMA_BF16(a, b, c) \
  __builtin_amdgcn_wmma_f32_16x16x32_bf16(false, (a), false, (b), (short)0, (c), false, false)

constexpr int   Bc = 4, Tc = 2048, Ec = 1024, Hc = 16, Sc = 64;
constexpr int   Mc = Bc * Tc;            // 8192 token rows
constexpr float SCALE = 0.125f;          // 1/sqrt(64)
constexpr float LN_EPS = 1e-5f;
constexpr float NEG_BIG = -1e30f;

// A-matrix (16xK, 16-bit) per-lane fragment: lanes 0-15 hold K=[0..15],
// lanes 16-31 hold K=[16..31] of each 32-wide K-step (contiguous, matches B).
__device__ inline v16bf load_a_frag(const bf16* row, int hi) {
  v8bf lo = *(const v8bf*)(row + hi * 16);
  v8bf h2 = *(const v8bf*)(row + hi * 16 + 8);
  v16bf r;
#pragma unroll
  for (int i = 0; i < 8; ++i) { r[i] = lo[i]; r[i + 8] = h2[i]; }
  return r;
}

// Async global -> LDS copy, 16 bytes per lane (tracked by ASYNCcnt).
__device__ inline void async_copy_b128(uint32_t lds_addr, const void* gaddr) {
  asm volatile("global_load_async_to_lds_b128 %0, %1, off"
               :: "v"(lds_addr), "v"(gaddr) : "memory");
}
__device__ inline void wait_async0() {
  asm volatile("s_wait_asynccnt 0" ::: "memory");
}

// ---------------- fp32 -> bf16 pack ----------------
__global__ __launch_bounds__(256) void cvt_bf16_kernel(const float* __restrict__ src,
                                                       bf16* __restrict__ dst, int n4) {
  int i = blockIdx.x * 256 + threadIdx.x;
  if (i >= n4) return;
  const float4 f = ((const float4*)src)[i];
  v4bf o;
  o[0] = (bf16)f.x; o[1] = (bf16)f.y; o[2] = (bf16)f.z; o[3] = (bf16)f.w;
  *(v4bf*)(dst + (size_t)i * 4) = o;
}

// ---------------- fused QKV projection + per-head LayerNorm ----------------
// Wave tile: 16 rows (tokens) x 64 cols (one head). Q/K -> (B,H,T,S), V -> (B,H,S,T).
__global__ __launch_bounds__(128) void qkv_ln_kernel(
    const bf16* __restrict__ xb,
    const bf16* __restrict__ Wqb, const bf16* __restrict__ Wkb, const bf16* __restrict__ Wvb,
    const float* __restrict__ qgam, const float* __restrict__ qbet,
    const float* __restrict__ kgam, const float* __restrict__ kbet,
    bf16* __restrict__ Qo, bf16* __restrict__ Ko, bf16* __restrict__ Vt) {
  const int lane = threadIdx.x & 31;
  const int wave = threadIdx.x >> 5;
  const int lr = lane & 15;
  const int hi = lane >> 4;
  const int m0 = blockIdx.x * 64 + wave * 16;
  const int h = blockIdx.y;
  const int n0 = h * Sc;
  const int bb = m0 >> 11;             // batch (T == 2048)
  const int tl = m0 & (Tc - 1);        // token within batch for tile row 0

  const bf16* xrow = xb + (size_t)(m0 + lr) * Ec;

#pragma unroll
  for (int mat = 0; mat < 3; ++mat) {
    const bf16* W = (mat == 0) ? Wqb : (mat == 1) ? Wkb : Wvb;
    const bf16* brow = W + (size_t)(n0 + lr) * Ec;   // y = x @ W^T : B[k][n] = W[n][k]
    v8f c0 = {}, c1 = {}, c2 = {}, c3 = {};
    for (int kk = 0; kk < Ec; kk += 32) {
      v16bf a  = load_a_frag(xrow + kk, hi);
      v16bf b0 = *(const v16bf*)(brow + kk + hi * 16);
      v16bf b1 = *(const v16bf*)(brow + (size_t)16 * Ec + kk + hi * 16);
      v16bf b2 = *(const v16bf*)(brow + (size_t)32 * Ec + kk + hi * 16);
      v16bf b3 = *(const v16bf*)(brow + (size_t)48 * Ec + kk + hi * 16);
      c0 = WMMA_BF16(a, b0, c0);
      c1 = WMMA_BF16(a, b1, c1);
      c2 = WMMA_BF16(a, b2, c2);
      c3 = WMMA_BF16(a, b3, c3);
      __builtin_prefetch(xrow + kk + 256, 0, 1);
      __builtin_prefetch(brow + kk + 256, 0, 1);
    }

    if (mat < 2) {
      // LayerNorm over the 64 head dims of each row (16 lanes x 4 tiles hold one row).
      const float* g  = (mat == 0) ? qgam : kgam;
      const float* bt = (mat == 0) ? qbet : kbet;
      const float g0 = g[lr],      g1 = g[16 + lr],  g2 = g[32 + lr],  g3 = g[48 + lr];
      const float e0 = bt[lr],     e1 = bt[16 + lr], e2 = bt[32 + lr], e3 = bt[48 + lr];
#pragma unroll
      for (int r = 0; r < 8; ++r) {
        float sum = c0[r] + c1[r] + c2[r] + c3[r];
        float sq  = c0[r] * c0[r] + c1[r] * c1[r] + c2[r] * c2[r] + c3[r] * c3[r];
#pragma unroll
        for (int d = 1; d < 16; d <<= 1) {
          sum += __shfl_xor(sum, d, 32);
          sq  += __shfl_xor(sq, d, 32);
        }
        const float mu  = sum * (1.0f / 64.0f);
        const float var = sq * (1.0f / 64.0f) - mu * mu;
        const float inv = rsqrtf(var + LN_EPS);
        c0[r] = (c0[r] - mu) * inv * g0 + e0;
        c1[r] = (c1[r] - mu) * inv * g1 + e1;
        c2[r] = (c2[r] - mu) * inv * g2 + e2;
        c3[r] = (c3[r] - mu) * inv * g3 + e3;
      }
      bf16* dst = ((mat == 0) ? Qo : Ko) + ((size_t)(bb * Hc + h) * Tc + tl) * Sc;
#pragma unroll
      for (int r = 0; r < 8; ++r) {
        bf16* p = dst + (size_t)(r + hi * 8) * Sc;
        p[lr]      = (bf16)c0[r];
        p[16 + lr] = (bf16)c1[r];
        p[32 + lr] = (bf16)c2[r];
        p[48 + lr] = (bf16)c3[r];
      }
    } else {
      // V transposed per head: Vt[(bb*H+h)*S + s][t]; 8 consecutive t per lane -> b128 store.
      bf16* dst = Vt + ((size_t)(bb * Hc + h) * Sc) * Tc + tl + hi * 8;
      v8bf p0, p1, p2, p3;
#pragma unroll
      for (int r = 0; r < 8; ++r) {
        p0[r] = (bf16)c0[r]; p1[r] = (bf16)c1[r];
        p2[r] = (bf16)c2[r]; p3[r] = (bf16)c3[r];
      }
      *(v8bf*)(dst + (size_t)lr * Tc)        = p0;
      *(v8bf*)(dst + (size_t)(16 + lr) * Tc) = p1;
      *(v8bf*)(dst + (size_t)(32 + lr) * Tc) = p2;
      *(v8bf*)(dst + (size_t)(48 + lr) * Tc) = p3;
    }
  }
}

// ---------------- causal flash attention ----------------
// Block = 4 waves = 4 consecutive 16-query tiles of the SAME (b,h).
// K/V chunks (32 keys) are double-buffered in LDS via async global->LDS copies
// and shared by all 4 waves (4x traffic reduction vs per-wave streaming).
__global__ __launch_bounds__(128) void attn_kernel(
    const bf16* __restrict__ Qi, const bf16* __restrict__ Ki,
    const bf16* __restrict__ Vt, bf16* __restrict__ Ao) {
  __shared__ __align__(16) bf16 kbuf[2][32 * 64];   // [key 0..31][s 0..63]
  __shared__ __align__(16) bf16 vbuf[2][64 * 32];   // [s 0..63][key 0..31]
  __shared__ __align__(16) bf16 ptile[4][16 * 40];  // per-wave P transpose staging

  const int tid = threadIdx.x;
  const int lane = tid & 31;
  const int wave = tid >> 5;
  const int lr = lane & 15;
  const int hi = lane >> 4;
  const int bh = blockIdx.x >> 5;                  // / (T/64 == 32)
  const int tbase = (blockIdx.x & 31) << 6;        // 64-query block base
  const int t0 = tbase + wave * 16;                // this wave's query tile
  const bf16* Qh = Qi + (size_t)bh * Tc * Sc;
  const bf16* Kh = Ki + (size_t)bh * Tc * Sc;
  const bf16* Vh = Vt + (size_t)bh * Sc * Tc;
  bf16* lp = ptile[wave];

  // Stage one 32-key chunk: K rows 128B each (8 x b128), V rows 64B each (4 x b128).
  // 512 x b128 total, 4 per thread.
  auto stage = [&](int kb, int buf) {
#pragma unroll
    for (int rep = 0; rep < 2; ++rep) {
      const int e = tid + rep * 128;
      const int krow = e >> 3, kseg = e & 7;
      async_copy_b128((uint32_t)(uintptr_t)&kbuf[buf][krow * 64 + kseg * 8],
                      Kh + (size_t)(kb + krow) * Sc + kseg * 8);
      const int vrow = e >> 2, vseg = e & 3;
      async_copy_b128((uint32_t)(uintptr_t)&vbuf[buf][vrow * 32 + vseg * 8],
                      Vh + (size_t)vrow * Tc + kb + vseg * 8);
    }
  };

  const v16bf qa0 = load_a_frag(Qh + (size_t)(t0 + lr) * Sc, hi);        // s 0..31
  const v16bf qa1 = load_a_frag(Qh + (size_t)(t0 + lr) * Sc + 32, hi);   // s 32..63

  v8f o0 = {}, o1 = {}, o2 = {}, o3 = {};
  float mrow[8], lrow[8];
#pragma unroll
  for (int r = 0; r < 8; ++r) { mrow[r] = NEG_BIG; lrow[r] = 0.0f; }

  const int nch = (tbase >> 5) + 2;                // chunks covering keys <= tbase+63
  stage(0, 0);
  wait_async0();
  __syncthreads();

  for (int ci = 0; ci < nch; ++ci) {
    const int kb = ci << 5;
    if (ci + 1 < nch) stage(kb + 32, (ci + 1) & 1);

    const bf16* kbase = kbuf[ci & 1];
    const bf16* vbase = vbuf[ci & 1];

    v8f s0 = {}, s1 = {};
    {
      const bf16* kp0 = kbase + lr * 64 + hi * 16;          // keys kb..kb+15
      const bf16* kp1 = kbase + (16 + lr) * 64 + hi * 16;   // keys kb+16..kb+31
      const v16bf k00 = *(const v16bf*)(kp0);
      const v16bf k01 = *(const v16bf*)(kp0 + 32);
      const v16bf k10 = *(const v16bf*)(kp1);
      const v16bf k11 = *(const v16bf*)(kp1 + 32);
      s0 = WMMA_BF16(qa0, k00, s0);
      s0 = WMMA_BF16(qa1, k01, s0);
      s1 = WMMA_BF16(qa0, k10, s1);
      s1 = WMMA_BF16(qa1, k11, s1);
    }
    const bool needmask = (kb + 31 > t0);   // fully-masked chunks degrade to no-ops
#pragma unroll
    for (int r = 0; r < 8; ++r) {
      const int trow = t0 + r + hi * 8;
      float v0 = s0[r] * SCALE;
      float v1 = s1[r] * SCALE;
      if (needmask) {
        if (kb + lr > trow)      v0 = NEG_BIG;
        if (kb + 16 + lr > trow) v1 = NEG_BIG;
      }
      float mx = fmaxf(v0, v1);
#pragma unroll
      for (int d = 1; d < 16; d <<= 1) mx = fmaxf(mx, __shfl_xor(mx, d, 32));
      const float mnew = fmaxf(mrow[r], mx);
      const float corr = __expf(mrow[r] - mnew);
      const float p0 = __expf(v0 - mnew);
      const float p1 = __expf(v1 - mnew);
      float ps = p0 + p1;
#pragma unroll
      for (int d = 1; d < 16; d <<= 1) ps += __shfl_xor(ps, d, 32);
      lrow[r] = lrow[r] * corr + ps;
      mrow[r] = mnew;
      o0[r] *= corr; o1[r] *= corr; o2[r] *= corr; o3[r] *= corr;
      const int prow = r + hi * 8;                 // C-layout row -> LDS (A-layout source)
      lp[prow * 40 + lr]      = (bf16)p0;
      lp[prow * 40 + 16 + lr] = (bf16)p1;
    }
    asm volatile("s_wait_dscnt 0" ::: "memory");   // in-wave LDS transpose hand-off
    const v16bf pa = load_a_frag(lp + lr * 40, hi);
    const bf16* vp = vbase + lr * 32 + hi * 16;
    const v16bf vb0 = *(const v16bf*)(vp);
    const v16bf vb1 = *(const v16bf*)(vp + 16 * 32);
    const v16bf vb2 = *(const v16bf*)(vp + 32 * 32);
    const v16bf vb3 = *(const v16bf*)(vp + 48 * 32);
    o0 = WMMA_BF16(pa, vb0, o0);
    o1 = WMMA_BF16(pa, vb1, o1);
    o2 = WMMA_BF16(pa, vb2, o2);
    o3 = WMMA_BF16(pa, vb3, o3);

    wait_async0();        // this wave's stage of chunk ci+1 landed in LDS
    __syncthreads();      // all waves done: safe to consume ci+1 / overwrite ci's buffer
  }

  const int bb = bh >> 4;                          // / H
  const int h = bh & 15;
  bf16* dst = Ao + ((size_t)bb * Tc + t0) * Ec + h * Sc;   // (M, E) context layout
#pragma unroll
  for (int r = 0; r < 8; ++r) {
    const float inv = 1.0f / lrow[r];
    bf16* p = dst + (size_t)(r + hi * 8) * Ec;
    p[lr]      = (bf16)(o0[r] * inv);
    p[16 + lr] = (bf16)(o1[r] * inv);
    p[32 + lr] = (bf16)(o2[r] * inv);
    p[48 + lr] = (bf16)(o3[r] * inv);
  }
}

// ---------------- output projection: out = ctx @ Wu^T + bu (fp32 out) ----------------
__global__ __launch_bounds__(128) void proj_kernel(
    const bf16* __restrict__ Ain, const bf16* __restrict__ Wub,
    const float* __restrict__ bias, float* __restrict__ out) {
  const int lane = threadIdx.x & 31;
  const int wave = threadIdx.x >> 5;
  const int lr = lane & 15;
  const int hi = lane >> 4;
  const int m0 = blockIdx.x * 64 + wave * 16;
  const int n0 = blockIdx.y * 64;

  const bf16* arow = Ain + (size_t)(m0 + lr) * Ec;
  const bf16* brow = Wub + (size_t)(n0 + lr) * Ec;
  v8f c0 = {}, c1 = {}, c2 = {}, c3 = {};
  for (int kk = 0; kk < Ec; kk += 32) {
    v16bf a  = load_a_frag(arow + kk, hi);
    v16bf b0 = *(const v16bf*)(brow + kk + hi * 16);
    v16bf b1 = *(const v16bf*)(brow + (size_t)16 * Ec + kk + hi * 16);
    v16bf b2 = *(const v16bf*)(brow + (size_t)32 * Ec + kk + hi * 16);
    v16bf b3 = *(const v16bf*)(brow + (size_t)48 * Ec + kk + hi * 16);
    c0 = WMMA_BF16(a, b0, c0);
    c1 = WMMA_BF16(a, b1, c1);
    c2 = WMMA_BF16(a, b2, c2);
    c3 = WMMA_BF16(a, b3, c3);
    __builtin_prefetch(arow + kk + 256, 0, 1);
    __builtin_prefetch(brow + kk + 256, 0, 1);
  }
  const float u0 = bias[n0 + lr],      u1 = bias[n0 + 16 + lr];
  const float u2 = bias[n0 + 32 + lr], u3 = bias[n0 + 48 + lr];
#pragma unroll
  for (int r = 0; r < 8; ++r) {
    float* p = out + (size_t)(m0 + r + hi * 8) * Ec + n0;
    p[lr]      = c0[r] + u0;
    p[16 + lr] = c1[r] + u1;
    p[32 + lr] = c2[r] + u2;
    p[48 + lr] = c3[r] + u3;
  }
}

extern "C" void kernel_launch(void* const* d_in, const int* in_sizes, int n_in,
                              void* d_out, int out_size, void* d_ws, size_t ws_size,
                              hipStream_t stream) {
  (void)in_sizes; (void)n_in; (void)out_size; (void)ws_size;
  const float* x  = (const float*)d_in[0];
  const float* Wq = (const float*)d_in[1];
  const float* Wk = (const float*)d_in[2];
  const float* Wv = (const float*)d_in[3];
  const float* Wu = (const float*)d_in[4];
  const float* bu = (const float*)d_in[5];
  const float* qg = (const float*)d_in[6];
  const float* qb = (const float*)d_in[7];
  const float* kg = (const float*)d_in[8];
  const float* kb = (const float*)d_in[9];
  float* out = (float*)d_out;

  // Workspace layout (bf16 elems): xb | Wq | Wk | Wv | Wu | Q | K | Vt | ctx  = 88 MB total
  bf16* xb  = (bf16*)d_ws;
  bf16* Wqb = xb  + (size_t)Mc * Ec;
  bf16* Wkb = Wqb + (size_t)Ec * Ec;
  bf16* Wvb = Wkb + (size_t)Ec * Ec;
  bf16* Wub = Wvb + (size_t)Ec * Ec;
  bf16* Qb  = Wub + (size_t)Ec * Ec;
  bf16* Kb  = Qb  + (size_t)Mc * Ec;
  bf16* Vtb = Kb  + (size_t)Mc * Ec;
  bf16* Ctx = Vtb + (size_t)Mc * Ec;

  const int nx4 = Mc * Ec / 4;
  const int nw4 = Ec * Ec / 4;
  cvt_bf16_kernel<<<(nx4 + 255) / 256, 256, 0, stream>>>(x, xb, nx4);
  cvt_bf16_kernel<<<(nw4 + 255) / 256, 256, 0, stream>>>(Wq, Wqb, nw4);
  cvt_bf16_kernel<<<(nw4 + 255) / 256, 256, 0, stream>>>(Wk, Wkb, nw4);
  cvt_bf16_kernel<<<(nw4 + 255) / 256, 256, 0, stream>>>(Wv, Wvb, nw4);
  cvt_bf16_kernel<<<(nw4 + 255) / 256, 256, 0, stream>>>(Wu, Wub, nw4);

  qkv_ln_kernel<<<dim3(Mc / 64, Hc), 128, 0, stream>>>(xb, Wqb, Wkb, Wvb,
                                                       qg, qb, kg, kb, Qb, Kb, Vtb);
  attn_kernel<<<dim3(Bc * Hc * (Tc / 64)), 128, 0, stream>>>(Qb, Kb, Vtb, Ctx);
  proj_kernel<<<dim3(Mc / 64, Ec / 64), 128, 0, stream>>>(Ctx, Wub, bu, out);
}
